// PVQ_19095424598551
// MI455X (gfx1250) — matrix operand order repaired and checked by reading, not compiled, not run on hardware
//
#include <hip/hip_runtime.h>
#include <stdint.h>

#define N_ROWS   65536
#define DIMS     512
#define S_STAGES 8
#define K_CODES  1024
#define D_SUB    64
#define LPAD     72   // padded bf16 row stride in LDS (144 B) -> conflict-free b128 reads
#define ROWS_BLK 256  // rows per block (8 waves x 2 row-tiles x 16)

typedef __bf16    v16bf __attribute__((ext_vector_type(16)));
typedef float     v8f   __attribute__((ext_vector_type(8)));
typedef uint32_t  u32x4 __attribute__((ext_vector_type(4)));

struct U32x8 { u32x4 lo; u32x4 hi; };

static __device__ __forceinline__ v16bf load2(const void* p0, const void* p1) {
    U32x8 t;
    t.lo = *(const u32x4*)p0;
    t.hi = *(const u32x4*)p1;
    return __builtin_bit_cast(v16bf, t);
}

static __device__ __forceinline__ uint32_t pack_bf16(float a, float b) {
    uint32_t ia = __float_as_uint(a), ib = __float_as_uint(b);
    ia = ia + 0x7FFFu + ((ia >> 16) & 1u);   // round-to-nearest-even
    ib = ib + 0x7FFFu + ((ib >> 16) & 1u);
    return (ia >> 16) | (ib & 0xFFFF0000u);
}

// ---------------------------------------------------------------------------
// Zero the codebooks_used bitmap region of d_out (harness poisons once).
__global__ void pvq_zero_counts(int* __restrict__ counts) {
    int i = blockIdx.x * 256 + threadIdx.x;
    if (i < S_STAGES * K_CODES) counts[i] = 0;
}

// ---------------------------------------------------------------------------
// One thread per codeword: f32 -> packed bf16 codebook + ||c||^2.
__global__ void pvq_prep(const float* __restrict__ cb,
                         uint32_t* __restrict__ cbq,
                         float* __restrict__ cnorm) {
    int id = blockIdx.x * 256 + threadIdx.x;           // 0..8191 = s*K + k
    const float* src = cb + (size_t)id * D_SUB;
    uint32_t* dst = cbq + (size_t)id * 32;
    float ss = 0.f;
#pragma unroll
    for (int i = 0; i < 32; ++i) {
        float a = src[2 * i], b = src[2 * i + 1];
        ss += a * a + b * b;
        dst[i] = pack_bf16(a, b);
    }
    cnorm[id] = ss;
}

// ---------------------------------------------------------------------------
// Block: 256 rows x one stage x all 1024 codewords. 8 waves; each wave owns
// TWO 16-row tiles sharing one B fragment stream (double-buffered from LDS).
// Score = ||c||^2 - 2 x.c  (||x||^2 dropped: constant per row for argmin).
__global__ __launch_bounds__(256) void pvq_argmin(
    const float* __restrict__ x, const uint32_t* __restrict__ cbq,
    const float* __restrict__ cnorm, int* __restrict__ idxbuf,
    int* __restrict__ counts)
{
    extern __shared__ char smem[];
    // layout: [1024][72] bf16 codebook | [256][72] bf16 x tile | [1024] f32 cnorm
    uint16_t* xL  = (uint16_t*)(smem + K_CODES * LPAD * 2);
    float*    cnL = (float*)(smem + K_CODES * LPAD * 2 + ROWS_BLK * LPAD * 2);

    const int tid     = threadIdx.x;
    const int s       = blockIdx.y;
    const int rowBase = blockIdx.x * ROWS_BLK;

    // --- stage codebook (pre-packed bf16) -> padded LDS
    {
        const u32x4* src = (const u32x4*)(cbq + (size_t)s * K_CODES * 32);
        for (int c = tid; c < K_CODES; c += 256) {
            u32x4* dst = (u32x4*)(smem + (size_t)c * (LPAD * 2));
            const u32x4* sc = src + (size_t)c * 8;
#pragma unroll
            for (int i = 0; i < 8; ++i) dst[i] = sc[i];
        }
    }
    // --- x tile: 256 rows x 64 dims, convert f32 -> bf16 into padded LDS
    {
        const int r = tid;   // one full row per thread
        const float4* xr4 = (const float4*)(x + (size_t)(rowBase + r) * DIMS + s * D_SUB);
        uint32_t* dstrow = (uint32_t*)xL + r * (LPAD / 2);
#pragma unroll
        for (int i = 0; i < 8; ++i) {
            float4 p = xr4[2 * i], q = xr4[2 * i + 1];
            u32x4 w;
            w[0] = pack_bf16(p.x, p.y);
            w[1] = pack_bf16(p.z, p.w);
            w[2] = pack_bf16(q.x, q.y);
            w[3] = pack_bf16(q.z, q.w);
            *(u32x4*)(dstrow + 4 * i) = w;
        }
    }
    // --- ||c||^2
    for (int i = tid; i < K_CODES; i += 256)
        cnL[i] = cnorm[s * K_CODES + i];

    __syncthreads();

    const int wave = tid >> 5;
    const int lane = tid & 31;
    const int ln   = lane & 15;
    const int hi   = lane >> 4;

    // A fragments: two 16x64 row tiles held in registers for the whole sweep.
    // bf16 A 16x32 layout: lanes 0-15 hold K={0..7,16..23}, lanes 16-31 K={8..15,24..31}.
    const int klo = hi * 8;
    const uint16_t* xrowA = xL + (wave * 32 + ln) * LPAD;
    const uint16_t* xrowB = xrowA + 16 * LPAD;
    v16bf A0a = load2(xrowA + klo,      xrowA + 16 + klo);
    v16bf A1a = load2(xrowA + 32 + klo, xrowA + 48 + klo);
    v16bf A0b = load2(xrowB + klo,      xrowB + 16 + klo);
    v16bf A1b = load2(xrowB + 32 + klo, xrowB + 48 + klo);

    float best[2][8];
    int   bidx[2][8];
#pragma unroll
    for (int j = 0; j < 8; ++j) {
        best[0][j] = 3.4e38f; bidx[0][j] = 0;
        best[1][j] = 3.4e38f; bidx[1][j] = 0;
    }

    // bf16 B 32x16 layout: lanes 0-15 col N=lane K={0..7,16..23}, lanes 16-31 col N=lane-16 K={8..15,24..31}.
    const int kB = hi * 8;
    const uint16_t* cbL = (const uint16_t*)smem;

    // prime double buffer with tile 0
    const uint16_t* cr0 = cbL + ln * LPAD + kB;
    v16bf B0c = load2(cr0,      cr0 + 16);
    v16bf B1c = load2(cr0 + 32, cr0 + 48);
    float cnc = cnL[ln];

    for (int t = 0; t < 64; ++t) {
        // prefetch next tile's B fragments + cnorm (wraps; extra work is benign)
        const int tn = (t + 1) & 63;
        const uint16_t* crn = cbL + (tn * 16 + ln) * LPAD + kB;
        v16bf B0n = load2(crn,      crn + 16);
        v16bf B1n = load2(crn + 32, crn + 48);
        float cnn = cnL[tn * 16 + ln];

        v8f accA = {0.f, 0.f, 0.f, 0.f, 0.f, 0.f, 0.f, 0.f};
        v8f accB = {0.f, 0.f, 0.f, 0.f, 0.f, 0.f, 0.f, 0.f};
        accA = __builtin_amdgcn_wmma_f32_16x16x32_bf16(false, A0a, false, B0c,
                                                       (short)0, accA, false, false);
        accB = __builtin_amdgcn_wmma_f32_16x16x32_bf16(false, A0b, false, B0c,
                                                       (short)0, accB, false, false);
        accA = __builtin_amdgcn_wmma_f32_16x16x32_bf16(false, A1a, false, B1c,
                                                       (short)0, accA, false, false);
        accB = __builtin_amdgcn_wmma_f32_16x16x32_bf16(false, A1b, false, B1c,
                                                       (short)0, accB, false, false);

        const int n = t * 16 + ln;
#pragma unroll
        for (int j = 0; j < 8; ++j) {
            float sa = cnc - 2.0f * accA[j];
            float sb = cnc - 2.0f * accB[j];
            if (sa < best[0][j]) { best[0][j] = sa; bidx[0][j] = n; }
            if (sb < best[1][j]) { best[1][j] = sb; bidx[1][j] = n; }
        }
        B0c = B0n; B1c = B1n; cnc = cnn;
    }

    // argmin reduction across the 16 codeword-lanes of each half-wave
#pragma unroll
    for (int off = 1; off <= 8; off <<= 1) {
#pragma unroll
        for (int rt = 0; rt < 2; ++rt) {
#pragma unroll
            for (int j = 0; j < 8; ++j) {
                float ov = __shfl_xor(best[rt][j], off, 32);
                int   oi = __shfl_xor(bidx[rt][j], off, 32);
                if (ov < best[rt][j] ||
                    (ov == best[rt][j] && oi < bidx[rt][j])) {
                    best[rt][j] = ov; bidx[rt][j] = oi;
                }
            }
        }
    }

    if (ln == 0) {                       // lanes 0 (rows 0..7) and 16 (rows 8..15)
#pragma unroll
        for (int rt = 0; rt < 2; ++rt) {
#pragma unroll
            for (int j = 0; j < 8; ++j) {
                int row = rowBase + wave * 32 + rt * 16 + j + hi * 8;
                idxbuf[row * S_STAGES + s] = bidx[rt][j];
                counts[s * K_CODES + bidx[rt][j]] = 1;   // idempotent
            }
        }
    }
}

// ---------------------------------------------------------------------------
// One wave per row: gather q, row norms via shfl reduction, write nsvq (or q).
__global__ __launch_bounds__(256) void pvq_finish(
    const float* __restrict__ x, const float* __restrict__ rnd,
    const float* __restrict__ cb, const int* __restrict__ idxbuf,
    const int* __restrict__ train, float* __restrict__ out)
{
    const int tid  = threadIdx.x;
    const int row  = blockIdx.x * 8 + (tid >> 5);
    const int lane = tid & 31;
    const int s    = lane >> 2;            // 4 lanes per 64-dim stage
    const int sub  = (lane & 3) * 16;

    const float4* xr = (const float4*)(x   + (size_t)row * DIMS + lane * 16);
    const float4* rr = (const float4*)(rnd + (size_t)row * DIMS + lane * 16);
    const int ci = idxbuf[row * S_STAGES + s];
    const float4* qr = (const float4*)(cb + ((size_t)s * K_CODES + ci) * D_SUB + sub);

    float4 xv[4], rv[4], qv[4];
    float err = 0.f, rsq = 0.f;
#pragma unroll
    for (int i = 0; i < 4; ++i) {
        xv[i] = xr[i]; rv[i] = rr[i]; qv[i] = qr[i];
        float ex = xv[i].x - qv[i].x, ey = xv[i].y - qv[i].y;
        float ez = xv[i].z - qv[i].z, ew = xv[i].w - qv[i].w;
        err += ex * ex + ey * ey + ez * ez + ew * ew;
        rsq += rv[i].x * rv[i].x + rv[i].y * rv[i].y
             + rv[i].z * rv[i].z + rv[i].w * rv[i].w;
    }
#pragma unroll
    for (int off = 1; off < 32; off <<= 1) {
        err += __shfl_xor(err, off, 32);
        rsq += __shfl_xor(rsq, off, 32);
    }
    const float scale = sqrtf(err) / sqrtf(rsq) + 1e-12f;
    const int tm = *train;

    float4* op = (float4*)(out + (size_t)row * DIMS + lane * 16);
#pragma unroll
    for (int i = 0; i < 4; ++i) {
        float4 o;
        if (tm) {
            o.x = xv[i].x + scale * rv[i].x;
            o.y = xv[i].y + scale * rv[i].y;
            o.z = xv[i].z + scale * rv[i].z;
            o.w = xv[i].w + scale * rv[i].w;
        } else {
            o = qv[i];
        }
        op[i] = o;
    }
}

// ---------------------------------------------------------------------------
extern "C" void kernel_launch(void* const* d_in, const int* in_sizes, int n_in,
                              void* d_out, int out_size, void* d_ws, size_t ws_size,
                              hipStream_t stream) {
    const float* x     = (const float*)d_in[0];
    const float* rnd   = (const float*)d_in[1];
    const float* cb    = (const float*)d_in[2];
    const int*   train = (const int*)d_in[3];

    float* out    = (float*)d_out;
    int*   counts = (int*)((float*)d_out + (size_t)N_ROWS * DIMS);

    char* ws = (char*)d_ws;
    uint32_t* cbq   = (uint32_t*)ws;                                  // 1 MB packed bf16
    float*    cnorm = (float*)(ws + (size_t)8192 * 32 * 4);           // 32 KB
    int*      idxb  = (int*)(ws + (size_t)8192 * 32 * 4 + 8192 * 4);  // 2 MB indices

    pvq_zero_counts<<<32, 256, 0, stream>>>(counts);
    pvq_prep<<<32, 256, 0, stream>>>(cb, cbq, cnorm);

    const size_t smem = (size_t)K_CODES * LPAD * 2     // codebook bf16 (147,456)
                      + (size_t)ROWS_BLK * LPAD * 2    // x tile bf16   (36,864)
                      + (size_t)K_CODES * 4;           // cnorm f32     (4,096) => 188,416 B
    pvq_argmin<<<dim3(N_ROWS / ROWS_BLK, S_STAGES), 256, smem, stream>>>(
        x, cbq, cnorm, idxb, counts);

    pvq_finish<<<N_ROWS / 8, 256, 0, stream>>>(x, rnd, cb, idxb, train, out);
}
